// OutcomeModel_8151847928046
// MI455X (gfx1250) — matrix operand order repaired and checked by reading, not compiled, not run on hardware
//
#include <hip/hip_runtime.h>
#include <hip/hip_bf16.h>

// ---------------- problem constants (fixed by the reference) ----------------
#define NN      10000
#define EE      50000
#define ETOT    (EE + NN)       // edges + self loops = 60000
#define HH      8
#define CC      256
#define HC      2048            // H*C
#define NOUT    1024
#define NCLS    283
#define NCLS_PAD 320            // next multiple of 64

// ---------------- WMMA types ----------------
typedef __attribute__((ext_vector_type(16))) __bf16        v16bf;
typedef __attribute__((ext_vector_type(8)))  float         v8f;
typedef __attribute__((ext_vector_type(8)))  unsigned int  v8u;

union Frag { v8u u; v16bf bf; };

__device__ inline Frag load_frag(const unsigned short* p, int gap) {
    // two 16-byte loads -> 8 dwords -> 16 bf16
    Frag f;
    uint4 lo = *reinterpret_cast<const uint4*>(p);
    uint4 hi = *reinterpret_cast<const uint4*>(p + gap);
    f.u = (v8u){lo.x, lo.y, lo.z, lo.w, hi.x, hi.y, hi.z, hi.w};
    return f;
}

__device__ inline unsigned short f32_to_bf16(float f) {
    unsigned u = __float_as_uint(f);
    unsigned r = u + 0x7FFFu + ((u >> 16) & 1u);   // round-to-nearest-even
    return (unsigned short)(r >> 16);
}

// monotone float -> uint key (for atomic segment-max)
__device__ inline unsigned f2key(float f) {
    unsigned u = __float_as_uint(f);
    return (u & 0x80000000u) ? ~u : (u | 0x80000000u);
}
__device__ inline float key2f(unsigned k) {
    unsigned u = (k & 0x80000000u) ? (k ^ 0x80000000u) : ~k;
    return __uint_as_float(u);
}

// ---------------- tiny utility kernels ----------------
__global__ void fill_f32(float* p, float v, size_t n) {
    size_t i = (size_t)blockIdx.x * blockDim.x + threadIdx.x;
    if (i < n) p[i] = v;
}
__global__ void fill_u32(unsigned* p, unsigned v, size_t n) {
    size_t i = (size_t)blockIdx.x * blockDim.x + threadIdx.x;
    if (i < n) p[i] = v;
}
__global__ void cast_bf16(const float* __restrict__ in, unsigned short* __restrict__ out, size_t n) {
    size_t i = (size_t)blockIdx.x * blockDim.x + threadIdx.x;
    if (i < n) out[i] = f32_to_bf16(in[i]);
}
// W [K, Nc] fp32  ->  Wt [NcPad, K] bf16 (rows >= Nc zero-padded)
__global__ void transpose_cast(const float* __restrict__ W, unsigned short* __restrict__ Wt,
                               int K, int Nc, int NcPad) {
    size_t gid = (size_t)blockIdx.x * blockDim.x + threadIdx.x;
    if (gid >= (size_t)NcPad * K) return;
    int n = (int)(gid / K);
    int k = (int)(gid % K);
    float v = (n < Nc) ? W[(size_t)k * Nc + n] : 0.0f;
    Wt[gid] = f32_to_bf16(v);
}

// ---------------- bf16 WMMA GEMM ----------------
// C[M, ldc(cols)] = A[M,K](bf16) * Bt[NcPad,K](bf16)^T  (+bias), fp32 accumulate.
// Requires K % 64 == 0 (true for all call sites: 256 / 2048).
// Block = 128 threads = 4 waves; wave w owns rows [by*128 + w*32, +32),
// block owns cols [bx*64, +64). Per K-step of 32: 2 A-frags x 4 B-frags ->
// 8 WMMAs. K loop unrolled 2x with ping-pong fragment buffers: each buffer is
// written once and consumed a half-iteration later -> no register copies, no
// WMMA->VALU hazard nops, loads covered by a half-iteration of WMMA work.
__global__ __launch_bounds__(128)
void gemm_bf16_wmma(const unsigned short* __restrict__ A,
                    const unsigned short* __restrict__ Bt,
                    const float* __restrict__ bias,
                    float* __restrict__ C,
                    int M, int K, int ldc, int Nvalid) {
    const int lane = threadIdx.x & 31;
    const int wave = threadIdx.x >> 5;
    const int m0   = blockIdx.y * 128 + wave * 32;
    const int n0   = blockIdx.x * 64;
    const int half = lane >> 4;     // 0: lanes 0-15, 1: lanes 16-31
    const int lr   = lane & 15;
    const int kbA  = half * 8;      // A: VGPR0-3 K=0..7 / 8..15, VGPR4-7 K=16..23 / 24..31
    const int kbB  = half * 8;      // B: same K order as A: VGPR0-3 K=0..7 / 8..15, VGPR4-7 K=16..23 / 24..31

    // Clamp OOB rows to the last valid row: loads stay in-bounds and
    // unconditional; results for clamped rows are never stored.
    int r0 = m0 + lr;       if (r0 > M - 1) r0 = M - 1;
    int r1 = m0 + 16 + lr;  if (r1 > M - 1) r1 = M - 1;

    const unsigned short* Ap0 = A + (size_t)r0 * K + kbA;
    const unsigned short* Ap1 = A + (size_t)r1 * K + kbA;
    const unsigned short* Bp[4];
#pragma unroll
    for (int t = 0; t < 4; ++t)
        Bp[t] = Bt + (size_t)(n0 + t * 16 + lr) * K + kbB;

    v8f acc[2][4] = {};

    Frag a0[2], b0[4], a1[2], b1[4];

    // prologue: even-half fragments at k = 0
    a0[0] = load_frag(Ap0, 16);
    a0[1] = load_frag(Ap1, 16);
#pragma unroll
    for (int t = 0; t < 4; ++t) b0[t] = load_frag(Bp[t], 16);

    int k0 = 0;
    for (; k0 + 64 < K; k0 += 64) {
        // prefetch odd half (k0+32)
        a1[0] = load_frag(Ap0 + k0 + 32, 16);
        a1[1] = load_frag(Ap1 + k0 + 32, 16);
#pragma unroll
        for (int t = 0; t < 4; ++t) b1[t] = load_frag(Bp[t] + k0 + 32, 16);
        // compute even half
#pragma unroll
        for (int t = 0; t < 4; ++t) {
            acc[0][t] = __builtin_amdgcn_wmma_f32_16x16x32_bf16(
                false, a0[0].bf, false, b0[t].bf, (short)0, acc[0][t], false, false);
            acc[1][t] = __builtin_amdgcn_wmma_f32_16x16x32_bf16(
                false, a0[1].bf, false, b0[t].bf, (short)0, acc[1][t], false, false);
        }
        // prefetch next even half (k0+64)
        a0[0] = load_frag(Ap0 + k0 + 64, 16);
        a0[1] = load_frag(Ap1 + k0 + 64, 16);
#pragma unroll
        for (int t = 0; t < 4; ++t) b0[t] = load_frag(Bp[t] + k0 + 64, 16);
        // compute odd half
#pragma unroll
        for (int t = 0; t < 4; ++t) {
            acc[0][t] = __builtin_amdgcn_wmma_f32_16x16x32_bf16(
                false, a1[0].bf, false, b1[t].bf, (short)0, acc[0][t], false, false);
            acc[1][t] = __builtin_amdgcn_wmma_f32_16x16x32_bf16(
                false, a1[1].bf, false, b1[t].bf, (short)0, acc[1][t], false, false);
        }
    }

    // epilogue: k0 == K - 64
    a1[0] = load_frag(Ap0 + k0 + 32, 16);
    a1[1] = load_frag(Ap1 + k0 + 32, 16);
#pragma unroll
    for (int t = 0; t < 4; ++t) b1[t] = load_frag(Bp[t] + k0 + 32, 16);
#pragma unroll
    for (int t = 0; t < 4; ++t) {
        acc[0][t] = __builtin_amdgcn_wmma_f32_16x16x32_bf16(
            false, a0[0].bf, false, b0[t].bf, (short)0, acc[0][t], false, false);
        acc[1][t] = __builtin_amdgcn_wmma_f32_16x16x32_bf16(
            false, a0[1].bf, false, b0[t].bf, (short)0, acc[1][t], false, false);
    }
#pragma unroll
    for (int t = 0; t < 4; ++t) {
        acc[0][t] = __builtin_amdgcn_wmma_f32_16x16x32_bf16(
            false, a1[0].bf, false, b1[t].bf, (short)0, acc[0][t], false, false);
        acc[1][t] = __builtin_amdgcn_wmma_f32_16x16x32_bf16(
            false, a1[1].bf, false, b1[t].bf, (short)0, acc[1][t], false, false);
    }

#pragma unroll
    for (int t = 0; t < 4; ++t) {
        const int ncol = n0 + t * 16 + lr;
        if (ncol >= Nvalid) continue;
        const float bv = bias ? bias[ncol] : 0.0f;
#pragma unroll
        for (int mi = 0; mi < 2; ++mi) {
#pragma unroll
            for (int v = 0; v < 8; ++v) {
                const int row = m0 + mi * 16 + half * 8 + v;  // C layout: VGPR v -> M=v (+8 upper half)
                if (row < M) C[(size_t)row * ldc + ncol] = acc[mi][t][v] + bv;
            }
        }
    }
}

// ---------------- attention coefficients: a = sum_c h[n,h,c]*att[h,c] ----------------
// one wave per (node, head); wave-reduced dot of length 256
__global__ __launch_bounds__(256)
void attn_coef(const float* __restrict__ h, const float* __restrict__ att_s,
               const float* __restrict__ att_d, float* __restrict__ a_src,
               float* __restrict__ a_dst) {
    const int wg = blockIdx.x * 8 + (threadIdx.x >> 5);
    if (wg >= NN * HH) return;
    const int lane = threadIdx.x & 31;
    const int n  = wg / HH;
    const int hd = wg % HH;
    const float* hp = h + (size_t)n * HC + hd * CC;
    const float* sp = att_s + hd * CC;
    const float* dp = att_d + hd * CC;
    float ss = 0.f, sd = 0.f;
    for (int i = lane; i < CC; i += 32) {
        float v = hp[i];
        ss += v * sp[i];
        sd += v * dp[i];
    }
    for (int o = 16; o > 0; o >>= 1) {
        ss += __shfl_xor(ss, o, 32);
        sd += __shfl_xor(sd, o, 32);
    }
    if (lane == 0) { a_src[wg] = ss; a_dst[wg] = sd; }
}

// ---------------- edge pass 1: leaky(e), atomic segment-max over dst ----------------
__global__ void edge_scores(const long long* __restrict__ ei,
                            const float* __restrict__ a_src, const float* __restrict__ a_dst,
                            float* __restrict__ eraw, unsigned* __restrict__ mkey) {
    size_t gid = (size_t)blockIdx.x * blockDim.x + threadIdx.x;
    if (gid >= (size_t)ETOT * HH) return;
    const int e  = (int)(gid >> 3);
    const int hd = (int)(gid & 7);
    long long s, d;
    if (e < EE) { s = ei[e]; d = ei[EE + e]; } else { s = d = e - EE; }  // self loops
    float v = a_src[s * HH + hd] + a_dst[d * HH + hd];
    v = (v > 0.f) ? v : 0.2f * v;           // leaky_relu slope 0.2
    eraw[gid] = v;
    atomicMax(&mkey[d * HH + hd], f2key(v));
}

// ---------------- edge pass 2: exp(e - max), atomic segment-sum ----------------
__global__ void edge_exp(const long long* __restrict__ ei,
                         const unsigned* __restrict__ mkey,
                         float* __restrict__ eraw, float* __restrict__ denom) {
    size_t gid = (size_t)blockIdx.x * blockDim.x + threadIdx.x;
    if (gid >= (size_t)ETOT * HH) return;
    const int e  = (int)(gid >> 3);
    const int hd = (int)(gid & 7);
    long long d;
    if (e < EE) { d = ei[EE + e]; } else { d = e - EE; }
    const float m  = key2f(mkey[d * HH + hd]);
    const float ex = __expf(eraw[gid] - m);
    eraw[gid] = ex;
    atomicAdd(&denom[d * HH + hd], ex);
}

// ---------------- scatter aggregation: out[dst] += h[src]*alpha ----------------
__global__ __launch_bounds__(256)
void aggregate(const long long* __restrict__ ei, const float* __restrict__ h,
               const float* __restrict__ eexp, const float* __restrict__ denom,
               float* __restrict__ out) {
    const int e = blockIdx.x;
    long long s, d;
    if (e < EE) { s = ei[e]; d = ei[EE + e]; } else { s = d = e - EE; }
    const float* hp = h   + (size_t)s * HC;
    float*       op = out + (size_t)d * HC;
    for (int i = threadIdx.x; i < HC; i += 256) {
        const int hd = i >> 8;
        const float alpha = eexp[(size_t)e * HH + hd] / (denom[d * HH + hd] + 1e-16f);
        atomicAdd(&op[i], hp[i] * alpha);
    }
}

// ---------------- bias + leaky(0.01) + cast to bf16 ----------------
__global__ void bias_act_cast(const float* __restrict__ in, const float* __restrict__ b,
                              unsigned short* __restrict__ outb, size_t total) {
    size_t i = (size_t)blockIdx.x * blockDim.x + threadIdx.x;
    if (i >= total) return;
    const int c = (int)(i & (HC - 1));
    float v = in[i] + b[c];
    v = (v > 0.f) ? v : 0.01f * v;
    outb[i] = f32_to_bf16(v);
}

// ---------------- gather output nodes + bias + leaky + cast ----------------
__global__ __launch_bounds__(256)
void gather_act_cast(const float* __restrict__ in, const float* __restrict__ b,
                     const long long* __restrict__ nodes, unsigned short* __restrict__ outb) {
    const int r = blockIdx.x;
    const long long n = nodes[r];
    const float* ip = in + (size_t)n * HC;
    unsigned short* op = outb + (size_t)r * HC;
    for (int c = threadIdx.x; c < HC; c += 256) {
        float v = ip[c] + b[c];
        v = (v > 0.f) ? v : 0.01f * v;
        op[c] = f32_to_bf16(v);
    }
}

// ---------------- row softmax (283 cols), wave per row ----------------
__global__ __launch_bounds__(256)
void softmax_rows(const float* __restrict__ logits, float* __restrict__ dout) {
    const int row = blockIdx.x * 8 + (threadIdx.x >> 5);
    if (row >= NOUT) return;
    const int lane = threadIdx.x & 31;
    const float* lp = logits + (size_t)row * NCLS_PAD;
    float vals[9];
    float mx = -3.4e38f;
    int cnt = 0;
    for (int j = lane; j < NCLS; j += 32) { float v = lp[j]; vals[cnt++] = v; mx = fmaxf(mx, v); }
    for (int o = 16; o > 0; o >>= 1) mx = fmaxf(mx, __shfl_xor(mx, o, 32));
    float s = 0.f;
    cnt = 0;
    for (int j = lane; j < NCLS; j += 32) { float e = __expf(vals[cnt] - mx); vals[cnt] = e; s += e; ++cnt; }
    for (int o = 16; o > 0; o >>= 1) s += __shfl_xor(s, o, 32);
    const float inv = 1.0f / s;
    cnt = 0;
    for (int j = lane; j < NCLS; j += 32) dout[(size_t)row * NCLS + j] = vals[cnt++] * inv;
}

// ============================================================================
static inline size_t align_up(size_t x, size_t a) { return (x + a - 1) & ~(a - 1); }
static inline int ceil_div(long long a, long long b) { return (int)((a + b - 1) / b); }

extern "C" void kernel_launch(void* const* d_in, const int* in_sizes, int n_in,
                              void* d_out, int out_size, void* d_ws, size_t ws_size,
                              hipStream_t stream) {
    (void)in_sizes; (void)n_in; (void)out_size; (void)ws_size;

    const float*     x    = (const float*)d_in[0];
    const long long* ei   = (const long long*)d_in[1];   // [2, E] int64
    const long long* outn = (const long long*)d_in[2];   // [1024] int64
    const float*     W1   = (const float*)d_in[3];
    const float*     as1  = (const float*)d_in[4];
    const float*     ad1  = (const float*)d_in[5];
    const float*     b1   = (const float*)d_in[6];
    const float*     W2   = (const float*)d_in[7];
    const float*     as2  = (const float*)d_in[8];
    const float*     ad2  = (const float*)d_in[9];
    const float*     b2   = (const float*)d_in[10];
    const float*     Wp   = (const float*)d_in[11];
    const float*     bp   = (const float*)d_in[12];
    float* out = (float*)d_out;

    // ---- carve workspace ----
    char* p = (char*)d_ws;
    size_t off = 0;
    auto carve = [&](size_t bytes) -> char* {
        off = align_up(off, 256);
        char* r = p + off;
        off += bytes;
        return r;
    };
    unsigned short* xb     = (unsigned short*)carve((size_t)NN * CC * 2);        // x bf16
    unsigned short* Wt1    = (unsigned short*)carve((size_t)HC * CC * 2);        // W1^T bf16 [2048,256]
    unsigned short* Wt2    = (unsigned short*)carve((size_t)HC * HC * 2);        // W2^T bf16 [2048,2048]
    unsigned short* Wtp    = (unsigned short*)carve((size_t)NCLS_PAD * HC * 2);  // Wp^T bf16 [320,2048]
    float*          hbuf   = (float*)carve((size_t)NN * HC * 4);                 // gemm out (h1 then h2)
    float*          aggbuf = (float*)carve((size_t)NN * HC * 4);                 // aggregated out
    unsigned short* hactb  = (unsigned short*)carve((size_t)NN * HC * 2);        // activated bf16
    float*          a_src  = (float*)carve((size_t)NN * HH * 4);
    float*          a_dst  = (float*)carve((size_t)NN * HH * 4);
    unsigned*       mkey   = (unsigned*)carve((size_t)NN * HH * 4);
    float*          denom  = (float*)carve((size_t)NN * HH * 4);
    float*          eraw   = (float*)carve((size_t)ETOT * HH * 4);
    unsigned short* hselb  = (unsigned short*)carve((size_t)NOUT * HC * 2);
    float*          logits = (float*)carve((size_t)NOUT * NCLS_PAD * 4);

    const size_t nNH   = (size_t)NN * HH;
    const size_t nFeat = (size_t)NN * HC;
    const size_t nEdgH = (size_t)ETOT * HH;

    // ---- conversions (once per call; deterministic) ----
    cast_bf16<<<ceil_div((long long)NN * CC, 256), 256, 0, stream>>>(x, xb, (size_t)NN * CC);
    transpose_cast<<<ceil_div((long long)HC * CC, 256), 256, 0, stream>>>(W1, Wt1, CC, HC, HC);
    transpose_cast<<<ceil_div((long long)HC * HC, 256), 256, 0, stream>>>(W2, Wt2, HC, HC, HC);
    transpose_cast<<<ceil_div((long long)NCLS_PAD * HC, 256), 256, 0, stream>>>(Wp, Wtp, HC, NCLS, NCLS_PAD);

    const dim3 gemmBlk(128);
    const dim3 gemmL(HC / 64, ceil_div(NN, 128));      // 32 x 79

    // ================= layer 1 =================
    gemm_bf16_wmma<<<gemmL, gemmBlk, 0, stream>>>(xb, Wt1, nullptr, hbuf, NN, CC, HC, HC);
    attn_coef<<<ceil_div((long long)nNH, 8), 256, 0, stream>>>(hbuf, as1, ad1, a_src, a_dst);
    fill_u32<<<ceil_div((long long)nNH, 256), 256, 0, stream>>>(mkey, 0u, nNH);
    fill_f32<<<ceil_div((long long)nNH, 256), 256, 0, stream>>>(denom, 0.f, nNH);
    edge_scores<<<ceil_div((long long)nEdgH, 256), 256, 0, stream>>>(ei, a_src, a_dst, eraw, mkey);
    edge_exp<<<ceil_div((long long)nEdgH, 256), 256, 0, stream>>>(ei, mkey, eraw, denom);
    fill_f32<<<ceil_div((long long)nFeat, 256), 256, 0, stream>>>(aggbuf, 0.f, nFeat);
    aggregate<<<ETOT, 256, 0, stream>>>(ei, hbuf, eraw, denom, aggbuf);
    bias_act_cast<<<ceil_div((long long)nFeat, 256), 256, 0, stream>>>(aggbuf, b1, hactb, nFeat);

    // ================= layer 2 =================
    gemm_bf16_wmma<<<gemmL, gemmBlk, 0, stream>>>(hactb, Wt2, nullptr, hbuf, NN, HC, HC, HC);
    attn_coef<<<ceil_div((long long)nNH, 8), 256, 0, stream>>>(hbuf, as2, ad2, a_src, a_dst);
    fill_u32<<<ceil_div((long long)nNH, 256), 256, 0, stream>>>(mkey, 0u, nNH);
    fill_f32<<<ceil_div((long long)nNH, 256), 256, 0, stream>>>(denom, 0.f, nNH);
    edge_scores<<<ceil_div((long long)nEdgH, 256), 256, 0, stream>>>(ei, a_src, a_dst, eraw, mkey);
    edge_exp<<<ceil_div((long long)nEdgH, 256), 256, 0, stream>>>(ei, mkey, eraw, denom);
    fill_f32<<<ceil_div((long long)nFeat, 256), 256, 0, stream>>>(aggbuf, 0.f, nFeat);
    aggregate<<<ETOT, 256, 0, stream>>>(ei, hbuf, eraw, denom, aggbuf);
    gather_act_cast<<<NOUT, 256, 0, stream>>>(aggbuf, b2, outn, hselb);

    // ================= projection + softmax =================
    const dim3 gemmP(NCLS_PAD / 64, ceil_div(NOUT, 128));  // 5 x 8
    gemm_bf16_wmma<<<gemmP, gemmBlk, 0, stream>>>(hselb, Wtp, bp, logits, NOUT, HC, NCLS_PAD, NCLS);
    softmax_rows<<<ceil_div(NOUT, 8), 256, 0, stream>>>(logits, out);
}
